// NetVLADLoopDetector_64415919505510
// MI455X (gfx1250) — matrix operand order repaired and checked by reading, not compiled, not run on hardware
//
#include <hip/hip_runtime.h>
#include <hip/hip_bf16.h>

// ---------------------------------------------------------------------------
// Types for CDNA5 WMMA (gfx1250, wave32)
// ---------------------------------------------------------------------------
typedef __bf16 bf16x16 __attribute__((ext_vector_type(16)));
typedef float  f32x8   __attribute__((ext_vector_type(8)));

union FragU {
    unsigned short u[16];
    uint4          q[2];
    bf16x16        v;
};

__device__ inline unsigned short f2bf(float f) {
    unsigned int u = __float_as_uint(f);
    unsigned int r = (u + 0x7FFFu + ((u >> 16) & 1u)) >> 16;   // RNE
    return (unsigned short)r;
}

// A-matrix (16x32, 16-bit) per-lane K index for half h, lane-group g (ISA 7.12.2)
__device__ inline int a_k_of_h(int h, int g) {
    return ((h >> 3) << 4) + (g << 3) + (h & 7);
}
__device__ inline int a_k_const(int h) {      // g*8 folded out
    return ((h >> 3) << 4) + (h & 7);         // {0..7, 16..23}
}

__device__ inline f32x8 bwmma(const FragU& a, const FragU& b, f32x8 c) {
    return __builtin_amdgcn_wmma_f32_16x16x32_bf16(
        false, a.v, false, b.v, (short)0, c, false, false);
}

__device__ inline int clampi(int v, int lo, int hi) {
    return v < lo ? lo : (v > hi ? hi : v);
}

// NHWC A-fragment row load: two contiguous 16B runs (ci = +0..7, +16..23),
// zeroed branchlessly when out of bounds (address pre-clamped to valid).
__device__ inline void load_a_nhwc(FragU& a, const unsigned short* p, bool ok) {
    const uint4* q = (const uint4*)p;
    uint4 t0 = q[0], t1 = q[2];               // +0 bytes, +32 bytes
    a.q[0].x = ok ? t0.x : 0u;  a.q[0].y = ok ? t0.y : 0u;
    a.q[0].z = ok ? t0.z : 0u;  a.q[0].w = ok ? t0.w : 0u;
    a.q[1].x = ok ? t1.x : 0u;  a.q[1].y = ok ? t1.y : 0u;
    a.q[1].z = ok ? t1.z : 0u;  a.q[1].w = ok ? t1.w : 0u;
}

__device__ inline void load_b(FragU& b, const unsigned short* bp) {
    const uint4* q = (const uint4*)bp;
    b.q[0] = q[0];
    b.q[1] = q[1];
}

// ---------------------------------------------------------------------------
// Weight pack into WMMA B-fragment order.
//   frag[((kt*NT + nt)*32 + lane)*16 + h], K-in-tile = (h/8)*16 + (lane/16)*8 + h%8,
//   n = nt*16 + lane%16.
// K ordering: CIN>0 (conv): k = (kh*3+kw)*CIN + ci -> W col ci*9+kh*3+kw
//             CIN==0 (FC):  identity.
// ---------------------------------------------------------------------------
__global__ __launch_bounds__(256) void pack_b(const float* __restrict__ W,
                                              unsigned short* __restrict__ frag,
                                              int KREAL, int NT, int CIN, long total) {
    long e = (long)blockIdx.x * 256 + threadIdx.x;
    if (e >= total) return;
    int  h    = (int)(e & 15);
    int  lane = (int)((e >> 4) & 31);
    long q    = e >> 9;
    int  nt   = (int)(q % NT);
    long kt   = q / NT;
    int  k    = (int)(kt * 32) + ((h >> 3) << 4) + ((lane >> 4) << 3) + (h & 7);
    int  n    = nt * 16 + (lane & 15);
    float val = 0.0f;
    if (k < KREAL) {
        int korig;
        if (CIN > 0) {
            int rc = k / CIN;
            int ci = k - rc * CIN;
            korig  = ci * 9 + rc;
        } else {
            korig = k;
        }
        val = W[(size_t)n * KREAL + korig];
    }
    frag[e] = f2bf(val);
}

// ---------------------------------------------------------------------------
// Fused conv3x3 (stride 2, pad 1) + bias + ReLU + 2x2 maxpool, implicit GEMM.
// Activations (CIN>=32 path) are NHWC bf16 -> A fragment = 2x b128 per row.
// conv1 (CIN=3) reads the harness's f32 NCHW input with a scalar gather.
// Output written NHWC bf16 (lane stores coalesce across the 16 channels).
// ---------------------------------------------------------------------------
template <int CIN, int HIN, int COUT, int HOUTC, int NTC, typename TIN>
__global__ __launch_bounds__(256) void conv_pool(const TIN* __restrict__ x,
                                                 const float* __restrict__ bias,
                                                 const unsigned short* __restrict__ wfrag,
                                                 unsigned short* __restrict__ out) {
    constexpr int WIN   = HIN;
    constexpr int WOT   = HOUTC / 16;
    constexpr int HP    = HOUTC / 2;
    constexpr int NT    = COUT / 16;
    constexpr int NTG   = NT / NTC;
    constexpr int KREAL = CIN * 9;
    constexpr int KT    = (KREAL + 31) / 32;
    constexpr int TPC   = (CIN >= 32) ? (CIN / 32) : 1;

    int gw   = blockIdx.x * 8 + (threadIdx.x >> 5);
    int lane = threadIdx.x & 31;
    int ntg  = gw % NTG;  int t = gw / NTG;
    int wot  = t % WOT;   t /= WOT;
    int hp   = t % HP;    int n = t / HP;
    if (n >= 16) return;

    int m = lane & 15, g = lane >> 4;
    int wo = wot * 16 + m;

    f32x8 acc[NTC][2] = {};

    for (int kt = 0; kt < KT; ++kt) {
        FragU a0, a1;
        if constexpr (CIN >= 32) {
            int rc     = kt / TPC;                 // kh*3+kw, uniform per tile
            int kh     = rc / 3, kw = rc - kh * 3;
            int baseci = (kt % TPC) * 32 + g * 8;
            int wi  = 2 * wo - 1 + kw;
            int hi0 = 4 * hp - 1 + kh;
            int hi1 = hi0 + 2;
            bool wb  = (unsigned)wi  < (unsigned)WIN;
            bool ok0 = wb && ((unsigned)hi0 < (unsigned)HIN);
            bool ok1 = wb && ((unsigned)hi1 < (unsigned)HIN);
            int wic  = clampi(wi, 0, WIN - 1);
            int h0c  = clampi(hi0, 0, HIN - 1);
            int h1c  = clampi(hi1, 0, HIN - 1);
            const unsigned short* xa = (const unsigned short*)x;   // NHWC
            const unsigned short* p0 = xa + (((size_t)n * HIN + h0c) * WIN + wic) * CIN + baseci;
            const unsigned short* p1 = xa + (((size_t)n * HIN + h1c) * WIN + wic) * CIN + baseci;
            load_a_nhwc(a0, p0, ok0);
            load_a_nhwc(a1, p1, ok1);
        } else {
            // conv1: CIN=3, single K-tile, per-h decode (k = (kh*3+kw)*3 + ci)
            #pragma unroll
            for (int h = 0; h < 16; ++h) {
                int k = kt * 32 + a_k_of_h(h, g);
                unsigned short v0 = 0, v1 = 0;
                if (k < KREAL) {
                    int rc = k / CIN, ci = k - rc * CIN;
                    int kh = rc / 3, kw = rc - kh * 3;
                    int wi  = 2 * wo - 1 + kw;
                    int hi0 = 4 * hp - 1 + kh;
                    int hi1 = hi0 + 2;
                    bool wb = (unsigned)wi < (unsigned)WIN;
                    int wic = clampi(wi, 0, WIN - 1);
                    const TIN* base = x + ((size_t)n * CIN + ci) * HIN * WIN + wic;  // NCHW f32
                    float t0 = (float)base[(size_t)clampi(hi0, 0, HIN - 1) * WIN];
                    float t1 = (float)base[(size_t)clampi(hi1, 0, HIN - 1) * WIN];
                    v0 = (wb && (unsigned)hi0 < (unsigned)HIN) ? f2bf(t0) : (unsigned short)0;
                    v1 = (wb && (unsigned)hi1 < (unsigned)HIN) ? f2bf(t1) : (unsigned short)0;
                }
                a0.u[h] = v0; a1.u[h] = v1;
            }
        }

        #pragma unroll
        for (int ntc = 0; ntc < NTC; ++ntc) {
            int nt = ntg * NTC + ntc;
            FragU b;
            load_b(b, wfrag + (((size_t)kt * NT + nt) * 32 + lane) * 16);
            acc[ntc][0] = bwmma(a0, b, acc[ntc][0]);
            acc[ntc][1] = bwmma(a1, b, acc[ntc][1]);
        }
    }

    // Epilogue: bias + ReLU + 2x2 maxpool; NHWC stores (coalesce over lanes)
    int    wpb = wot * 8 + g * 4;
    size_t pix = ((size_t)n * HP + hp) * HP;     // NHWC row base (pixels)
    #pragma unroll
    for (int ntc = 0; ntc < NTC; ++ntc) {
        int   co  = (ntg * NTC + ntc) * 16 + m;
        float bvv = bias[co];
        #pragma unroll
        for (int j = 0; j < 4; ++j) {
            float qv = fmaxf(fmaxf(acc[ntc][0][2 * j], acc[ntc][0][2 * j + 1]),
                             fmaxf(acc[ntc][1][2 * j], acc[ntc][1][2 * j + 1]));
            out[(pix + wpb + j) * COUT + co] = f2bf(fmaxf(qv + bvv, 0.0f));
        }
    }
}

// ---------------------------------------------------------------------------
// conv4 (256->512, 8x8 -> 4x4, NHWC in) + bias + ReLU + fused global avg pool.
// ---------------------------------------------------------------------------
__global__ __launch_bounds__(256) void conv4_feats(const unsigned short* __restrict__ act3,
                                                   const float* __restrict__ bias,
                                                   const unsigned short* __restrict__ wfrag,
                                                   float* __restrict__ feats,
                                                   unsigned short* __restrict__ featsbf) {
    constexpr int NT = 32, NTC = 8, NTG = NT / NTC, KT = 72, TPC = 8; // CIN=256
    int gw = blockIdx.x * 8 + (threadIdx.x >> 5);
    int lane = threadIdx.x & 31;
    int ntg = gw % NTG, n = gw / NTG;
    if (n >= 16) return;
    int m = lane & 15, g = lane >> 4;
    int ho = m >> 2, wo = m & 3;

    f32x8 acc[NTC] = {};
    for (int kt = 0; kt < KT; ++kt) {
        int rc     = kt / TPC;
        int kh     = rc / 3, kw = rc - kh * 3;
        int baseci = (kt % TPC) * 32 + g * 8;
        int hi = 2 * ho - 1 + kh, wi = 2 * wo - 1 + kw;
        bool ok = ((unsigned)hi < 8u) && ((unsigned)wi < 8u);
        int hic = clampi(hi, 0, 7), wic = clampi(wi, 0, 7);
        const unsigned short* p = act3 + (((size_t)n * 8 + hic) * 8 + wic) * 256 + baseci;

        FragU a;
        load_a_nhwc(a, p, ok);
        #pragma unroll
        for (int ntc = 0; ntc < NTC; ++ntc) {
            int nt = ntg * NTC + ntc;
            FragU b;
            load_b(b, wfrag + (((size_t)kt * NT + nt) * 32 + lane) * 16);
            acc[ntc] = bwmma(a, b, acc[ntc]);
        }
    }

    #pragma unroll
    for (int ntc = 0; ntc < NTC; ++ntc) {
        int   co  = (ntg * NTC + ntc) * 16 + m;
        float bvv = bias[co];
        float s = 0.0f;
        #pragma unroll
        for (int v = 0; v < 8; ++v) s += fmaxf(acc[ntc][v] + bvv, 0.0f);
        s += __shfl_xor(s, 16, 32);
        float f = s * (1.0f / 16.0f);
        if (lane < 16) {
            feats[(size_t)n * 512 + co]   = f;
            featsbf[(size_t)n * 512 + co] = f2bf(f);
        }
    }
}

// ---------------------------------------------------------------------------
// logits = feats @ wv^T + bv (WMMA, 4 waves) fused with row softmax via LDS.
// ---------------------------------------------------------------------------
__global__ __launch_bounds__(128) void logits_softmax(const unsigned short* __restrict__ featsbf,
                                                      const unsigned short* __restrict__ wvfrag,
                                                      const float* __restrict__ bv,
                                                      float* __restrict__ aout) {
    __shared__ float lg[16][64];
    int lane = threadIdx.x & 31, w = threadIdx.x >> 5;   // w = ntile (0..3)
    int m = lane & 15, g = lane >> 4;

    f32x8 c = {};
    for (int kt = 0; kt < 16; ++kt) {
        FragU a, b;
        const unsigned short* ap = featsbf + (size_t)m * 512 + kt * 32 + g * 8;
        #pragma unroll
        for (int h = 0; h < 16; ++h) a.u[h] = ap[a_k_const(h)];
        load_b(b, wvfrag + (((size_t)kt * 4 + w) * 32 + lane) * 16);
        c = bwmma(a, b, c);
    }
    float bvv = bv[w * 16 + m];
    #pragma unroll
    for (int v = 0; v < 8; ++v) lg[v + 8 * g][w * 16 + m] = c[v] + bvv;
    __syncthreads();

    if (threadIdx.x < 16) {
        int r = threadIdx.x;
        float mx = -3.0e38f;
        for (int k = 0; k < 64; ++k) mx = fmaxf(mx, lg[r][k]);
        float s = 0.0f;
        for (int k = 0; k < 64; ++k) s += expf(lg[r][k] - mx);
        float inv = 1.0f / s;
        for (int k = 0; k < 64; ++k) aout[r * 64 + k] = expf(lg[r][k] - mx) * inv;
    }
}

// ---------------------------------------------------------------------------
// vlad[b][k][c] = a[b][k] * (feats[b][c] - centroids[k][c])  -> bf16
// ---------------------------------------------------------------------------
__global__ __launch_bounds__(256) void vlad_kernel(const float* __restrict__ a,
                                                   const float* __restrict__ feats,
                                                   const float* __restrict__ cent,
                                                   unsigned short* __restrict__ vlad) {
    int idx = blockIdx.x * 256 + threadIdx.x;   // 16*64*512 = 524288 exact
    int c = idx & 511;
    int k = (idx >> 9) & 63;
    int b = idx >> 15;
    float v = a[b * 64 + k] * (feats[b * 512 + c] - cent[k * 512 + c]);
    vlad[idx] = f2bf(v);
}

// ---------------------------------------------------------------------------
// Generic M=16 FC GEMM with optional split-K.
// splitK==1: out = A x B + bias [,ReLU] -> outbf/outf
// splitK >1: partial K-chunk sums (no bias) -> outf[(blockIdx.y*16+row)*N+col]
// ---------------------------------------------------------------------------
template <int NTC>
__global__ __launch_bounds__(32) void fc_gemm(const unsigned short* __restrict__ A,
                                              const unsigned short* __restrict__ bfrag,
                                              const float* __restrict__ bias,
                                              int Kdim, int NT, int relu, int splitK,
                                              unsigned short* __restrict__ outbf,
                                              float* __restrict__ outf) {
    int lane = threadIdx.x & 31;
    int ntg = blockIdx.x;
    int s   = blockIdx.y;
    int m = lane & 15, g = lane >> 4;
    int KT  = Kdim >> 5;
    int ktc = KT / splitK;
    int kt0 = s * ktc;

    f32x8 acc[NTC] = {};
    for (int kt = kt0; kt < kt0 + ktc; ++kt) {
        FragU a;
        const unsigned short* ap = A + (size_t)m * Kdim + kt * 32 + g * 8;
        #pragma unroll
        for (int h = 0; h < 16; ++h) a.u[h] = ap[a_k_const(h)];
        #pragma unroll
        for (int ntc = 0; ntc < NTC; ++ntc) {
            int nt = ntg * NTC + ntc;
            FragU b;
            const unsigned short* bp = bfrag + (((size_t)kt * NT + nt) * 32 + lane) * 16;
            __builtin_prefetch((const void*)(bp + (size_t)NT * 512), 0, 3);
            load_b(b, bp);
            acc[ntc] = bwmma(a, b, acc[ntc]);
        }
    }

    int N = NT * 16;
    #pragma unroll
    for (int ntc = 0; ntc < NTC; ++ntc) {
        int col = (ntg * NTC + ntc) * 16 + m;
        #pragma unroll
        for (int v = 0; v < 8; ++v) {
            int row = v + 8 * g;
            if (splitK > 1) {
                outf[((size_t)s * 16 + row) * N + col] = acc[ntc][v];
            } else {
                float val = acc[ntc][v] + bias[col];
                if (relu) val = fmaxf(val, 0.0f);
                if (outbf) outbf[(size_t)row * N + col] = f2bf(val);
                if (outf)  outf[(size_t)row * N + col]  = val;
            }
        }
    }
}

// Reduce split-K partials, add bias, ReLU, emit bf16.
__global__ __launch_bounds__(256) void fc_reduce(const float* __restrict__ part,
                                                 const float* __restrict__ bias,
                                                 unsigned short* __restrict__ outbf,
                                                 int N, int splitK) {
    int idx = blockIdx.x * 256 + threadIdx.x;   // 16*N
    int col = idx % N, row = idx / N;
    float s = 0.0f;
    for (int k = 0; k < splitK; ++k) s += part[((size_t)k * 16 + row) * N + col];
    s = fmaxf(s + bias[col], 0.0f);
    outbf[idx] = f2bf(s);
}

// ---------------------------------------------------------------------------
// Per-row L2 normalization (16 rows x 256)
// ---------------------------------------------------------------------------
__global__ __launch_bounds__(256) void l2norm(const float* __restrict__ dpre,
                                              float* __restrict__ out) {
    __shared__ float red[256];
    int r = blockIdx.x, t = threadIdx.x;
    float v = dpre[r * 256 + t];
    red[t] = v * v;
    __syncthreads();
    for (int s = 128; s > 0; s >>= 1) {
        if (t < s) red[t] += red[t + s];
        __syncthreads();
    }
    out[r * 256 + t] = v * rsqrtf(red[0]);
}

// ---------------------------------------------------------------------------
// Host-side launch
// ---------------------------------------------------------------------------
extern "C" void kernel_launch(void* const* d_in, const int* in_sizes, int n_in,
                              void* d_out, int out_size, void* d_ws, size_t ws_size,
                              hipStream_t stream) {
    const float* x    = (const float*)d_in[0];
    const float* w1   = (const float*)d_in[1];  const float* b1  = (const float*)d_in[2];
    const float* w2   = (const float*)d_in[3];  const float* b2  = (const float*)d_in[4];
    const float* w3   = (const float*)d_in[5];  const float* b3  = (const float*)d_in[6];
    const float* w4   = (const float*)d_in[7];  const float* b4  = (const float*)d_in[8];
    const float* wv   = (const float*)d_in[9];  const float* bv  = (const float*)d_in[10];
    const float* cent = (const float*)d_in[11];
    const float* fw1  = (const float*)d_in[12]; const float* fb1 = (const float*)d_in[13];
    const float* fw2  = (const float*)d_in[14]; const float* fb2 = (const float*)d_in[15];
    float* out = (float*)d_out;

    char*  ws  = (char*)d_ws;
    size_t off = 0;
    auto alloc = [&](size_t bytes) -> void* {
        void* p = ws + off;
        off += (bytes + 255) & ~(size_t)255;
        return p;
    };

    // activations (NHWC bf16)
    unsigned short* act1    = (unsigned short*)alloc((size_t)16 * 128 * 128 * 64 * 2);
    unsigned short* act2    = (unsigned short*)alloc((size_t)16 * 32 * 32 * 128 * 2);
    unsigned short* act3    = (unsigned short*)alloc((size_t)16 * 8 * 8 * 256 * 2);
    float*          feats   = (float*)alloc(16 * 512 * 4);
    unsigned short* featsbf = (unsigned short*)alloc(16 * 512 * 2);
    float*          asoft   = (float*)alloc(16 * 64 * 4);
    unsigned short* vlad    = (unsigned short*)alloc((size_t)16 * 64 * 512 * 2);
    float*          part1   = (float*)alloc((size_t)32 * 16 * 256 * 4);   // fc1 split-K partials
    unsigned short* hidden  = (unsigned short*)alloc(16 * 256 * 2);
    float*          dpre    = (float*)alloc(16 * 256 * 4);
    // packed weight fragments  (KT * NT * 512 elements each)
    unsigned short* w1f  = (unsigned short*)alloc((size_t)1    * 4  * 512 * 2);
    unsigned short* w2f  = (unsigned short*)alloc((size_t)18   * 8  * 512 * 2);
    unsigned short* w3f  = (unsigned short*)alloc((size_t)36   * 16 * 512 * 2);
    unsigned short* w4f  = (unsigned short*)alloc((size_t)72   * 32 * 512 * 2);
    unsigned short* wvf  = (unsigned short*)alloc((size_t)16   * 4  * 512 * 2);
    unsigned short* fw1f = (unsigned short*)alloc((size_t)1024 * 16 * 512 * 2);
    unsigned short* fw2f = (unsigned short*)alloc((size_t)8    * 16 * 512 * 2);

    auto packGrid = [](long total) { return (unsigned)((total + 255) / 256); };

    // --- pack all weights into WMMA B-fragment order (bf16) ---
    pack_b<<<packGrid(2048),    256, 0, stream>>>(w1,  w1f,  27,    4,  3,   2048);
    pack_b<<<packGrid(73728),   256, 0, stream>>>(w2,  w2f,  576,   8,  64,  73728);
    pack_b<<<packGrid(294912),  256, 0, stream>>>(w3,  w3f,  1152,  16, 128, 294912);
    pack_b<<<packGrid(1179648), 256, 0, stream>>>(w4,  w4f,  2304,  32, 256, 1179648);
    pack_b<<<packGrid(32768),   256, 0, stream>>>(wv,  wvf,  512,   4,  0,   32768);
    pack_b<<<packGrid(8388608), 256, 0, stream>>>(fw1, fw1f, 32768, 16, 0,   8388608);
    pack_b<<<packGrid(65536),   256, 0, stream>>>(fw2, fw2f, 256,   16, 0,   65536);

    // --- backbone: fused conv + ReLU + maxpool (NHWC bf16 activations) ---
    // waves = 16 * HP * (HOUTC/16) * (NT/NTC); 8 waves per 256-thread block
    conv_pool<3,   512, 64,  256, 4, float>         <<<4096, 256, 0, stream>>>(x, b1, w1f, act1);
    conv_pool<64,  128, 128, 64,  8, unsigned short><<<256,  256, 0, stream>>>(act1, b2, w2f, act2);
    conv_pool<128, 32,  256, 16,  8, unsigned short><<<32,   256, 0, stream>>>(act2, b3, w3f, act3);
    conv4_feats<<<8, 256, 0, stream>>>(act3, b4, w4f, feats, featsbf);

    // --- NetVLAD head ---
    logits_softmax<<<1, 128, 0, stream>>>(featsbf, wvf, bv, asoft);
    vlad_kernel<<<2048, 256, 0, stream>>>(asoft, feats, cent, vlad);

    // --- descriptor MLP + L2 norm ---
    fc_gemm<4><<<dim3(4, 32), 32, 0, stream>>>(vlad, fw1f, fb1, 32768, 16, 1, 32,
                                               (unsigned short*)nullptr, part1);
    fc_reduce<<<16, 256, 0, stream>>>(part1, fb1, hidden, 256, 32);
    fc_gemm<4><<<dim3(4, 1), 32, 0, stream>>>(hidden, fw2f, fb2, 256, 16, 0, 1,
                                              (unsigned short*)nullptr, dpre);
    l2norm<<<16, 256, 0, stream>>>(dpre, out);

    (void)in_sizes; (void)n_in; (void)out_size; (void)ws_size;
}
